// ModConv2d_43885975831074
// MI455X (gfx1250) — matrix operand order repaired and checked by reading, not compiled, not run on hardware
//
#include <hip/hip_runtime.h>
#include <hip/hip_bf16.h>

// ---------------------------------------------------------------------------
// StyleGAN2 modulated conv2d for MI455X (gfx1250), wave32 + WMMA + async-LDS.
//   BS=8, C_IN=256, C_OUT=256, STYLE_D=512, H=W=64, F=3
// Implicit GEMM: per-sample  Out[256,4096] = A[256,2304] x B[2304,4096]
//   k ordered as k = r*256 + ci   (r = fy*3+fx), so each filter tap gives a
//   contiguous k-block of 32 -> one V_WMMA_F32_16X16X32_F16 step.
// The im2col x-window is staged into LDS with GLOBAL_LOAD_ASYNC_TO_LDS_B128
// (ASYNCcnt-tracked DMA), double-buffered so the DMA of chunk cb+1 overlaps
// the WMMA work of chunk cb.
// ---------------------------------------------------------------------------

typedef __attribute__((ext_vector_type(16))) _Float16 v16h;
typedef __attribute__((ext_vector_type(8)))  _Float16 v8h;
typedef __attribute__((ext_vector_type(8)))  float    v8f;

#define BS      8
#define C_IN    256
#define C_OUT   256
#define STYLE_D 512
#define HH      64
#define WW      64
#define KTOT    2304        // 9 * 256
#define NPIX    4096        // 64 * 64
#define XWIN    (4 * 66 * 32)   // padded window: 4 rows x 66 cols x 32 ci

// ---------------------------------------------------------------------------
// Kernel 1: style1[b][ci] = s[b] . affine_w[ci] + affine_b[ci] + 1
// ---------------------------------------------------------------------------
__global__ __launch_bounds__(256)
void style_kernel(const float* __restrict__ s,
                  const float* __restrict__ aw,
                  const float* __restrict__ ab,
                  float* __restrict__ style1)
{
    int idx = blockIdx.x * 256 + threadIdx.x;        // 0 .. 2047
    int b  = idx >> 8;
    int ci = idx & 255;
    const float* sp = s  + b  * STYLE_D;
    const float* wp = aw + ci * STYLE_D;
    float acc = ab[ci];
    for (int k = 0; k < STYLE_D; ++k)
        acc = fmaf(sp[k], wp[k], acc);
    style1[idx] = acc + 1.0f;
}

// ---------------------------------------------------------------------------
// Kernel 2: modulate + demodulate, pack A-matrix as f16:
//   Amod[b][co][r*256+ci] = w[co][ci][r] * style1[b][ci] * rsqrt(sumsq+1e-8)
// One block (256 thr) per (b, co). Weight row is staged in LDS (coalesced
// global read); Amod is written in linear output order (coalesced stores).
// ---------------------------------------------------------------------------
__global__ __launch_bounds__(256)
void modw_kernel(const float* __restrict__ w,
                 const float* __restrict__ style1,
                 _Float16* __restrict__ Amod)
{
    __shared__ float wrow[KTOT];
    __shared__ float red[256];
    int b  = blockIdx.x >> 8;
    int co = blockIdx.x & 255;
    int t  = threadIdx.x;

    for (int e = t; e < KTOT; e += 256)              // coalesced
        wrow[e] = w[(size_t)co * KTOT + e];
    __syncthreads();

    float wm[9];
    float part = 0.0f;
    for (int i = 0; i < 9; ++i) {
        int j  = t + i * 256;            // output-linear index: r*256 + ci
        int ci = j & 255;
        int r  = j >> 8;
        float v = wrow[ci * 9 + r] * style1[b * C_IN + ci];
        wm[i] = v;
        part  = fmaf(v, v, part);
    }
    red[t] = part;
    __syncthreads();
    for (int sft = 128; sft > 0; sft >>= 1) {
        if (t < sft) red[t] += red[t + sft];
        __syncthreads();
    }
    float inv = rsqrtf(red[0] + 1e-8f);

    _Float16* dst = Amod + (size_t)(b * C_OUT + co) * KTOT;
    for (int i = 0; i < 9; ++i)                       // coalesced f16 stores
        dst[t + i * 256] = (_Float16)(wm[i] * inv);
}

// ---------------------------------------------------------------------------
// Kernel 3: NCHW f32 -> NHWC f16 via LDS tile transpose.
//   Tile: 64 pixels x 32 channels. Reads coalesced over hw, writes coalesced
//   over ci (so the GEMM's async B copies are 16B-aligned contiguous runs).
// ---------------------------------------------------------------------------
__global__ __launch_bounds__(256)
void xcvt_kernel(const float* __restrict__ x, _Float16* __restrict__ Xh)
{
    __shared__ float tile[64][33];
    int hwt = blockIdx.x & 63;            // 64 hw tiles of 64 pixels
    int cit = (blockIdx.x >> 6) & 7;      // 8 ci tiles of 32 channels
    int b   = blockIdx.x >> 9;            // 8 samples
    int hw0 = hwt * 64, ci0 = cit * 32;
    int t = threadIdx.x;

    for (int i = 0; i < 8; ++i) {
        int idx  = t + i * 256;           // 0..2047
        int hw_l = idx & 63;
        int ci_l = idx >> 6;              // 0..31
        tile[hw_l][ci_l] = x[(size_t)(b * C_IN + ci0 + ci_l) * NPIX + hw0 + hw_l];
    }
    __syncthreads();
    for (int i = 0; i < 8; ++i) {
        int idx  = t + i * 256;
        int ci_l = idx & 31;
        int hw_l = idx >> 5;
        Xh[(size_t)(b * NPIX + hw0 + hw_l) * C_IN + ci0 + ci_l] = (_Float16)tile[hw_l][ci_l];
    }
}

// ---------------------------------------------------------------------------
// Kernel 4: main implicit-GEMM conv with WMMA + async double-buffered LDS.
//   grid = (32 n-tiles, 4 m-tiles, 8 samples), block = 256 thr = 8 wave32.
//   block tile: M=64 (c_out) x N=128 (2 image rows).
//   wave tile : 32x32 -> 2x2 sub-tiles of 16x16, 4 x v8f accumulators.
//   LDS: 2 x padded x window (32ch x 4rows x 66cols f16 = 16,896 B each).
//   Borders / out-of-image rows pre-zeroed once; async DMA fills interior.
// ---------------------------------------------------------------------------
__global__ __launch_bounds__(256)
void modconv_gemm(const _Float16* __restrict__ Amod,
                  const _Float16* __restrict__ Xh,
                  const float* __restrict__ bias,
                  float* __restrict__ out)
{
    __shared__ __attribute__((aligned(64))) _Float16 xs[2][XWIN];

    const int nt   = blockIdx.x;             // 0..31  (pixel tile: 2 rows)
    const int mt   = blockIdx.y;             // 0..3   (c_out tile of 64)
    const int b    = blockIdx.z;             // 0..7
    const int tid  = threadIdx.x;
    const int lane = tid & 31;
    const int wave = tid >> 5;
    const int wm   = wave >> 2;              // 0..1  (M position)
    const int wn   = wave & 3;               // 0..3  (N position)
    const int lhi  = lane >> 4;              // 0/1 : lane group
    const int l16  = lane & 15;
    const int h0   = nt * 2;                 // first output row of this tile

    // Zero both buffers once: border columns (0,65) and out-of-image rows are
    // never touched by the async DMA, so they stay zero for every chunk.
    for (int e = tid; e < 2 * XWIN; e += 256)
        ((_Float16*)xs)[e] = (_Float16)0.0f;
    __syncthreads();

    // Issue async DMA of chunk cb's interior x-window into buffer `buf`.
    // 1024 slots of 16B: 4 rows x 64 cols x 4 groups (8 halves each).
    auto issue = [&](int cb, int buf) {
        const int ci0 = cb * 32;
        for (int e = tid; e < 1024; e += 256) {        // 4 per thread
            int grp = e & 3;
            int rc  = e >> 2;
            int win = rc & 63;
            int row = rc >> 6;                          // 0..3
            int hin = h0 - 1 + row;
            if ((unsigned)hin < (unsigned)HH) {
                unsigned lds = (unsigned)(size_t)&xs[buf][(row * 66 + (win + 1)) * 32 + grp * 8];
                unsigned long long g = (unsigned long long)(size_t)
                    (Xh + ((size_t)((b * HH + hin) * WW + win)) * C_IN + ci0 + grp * 8);
                asm volatile("global_load_async_to_lds_b128 %0, %1, off"
                             :: "v"(lds), "v"(g) : "memory");
            }
        }
    };

    v8f acc[2][2] = {};

    issue(0, 0);

    for (int cb = 0; cb < 8; ++cb) {
        // Own async copies for chunk cb complete, then block-wide barrier:
        // after it, every wave's chunk-cb data is visible in xs[cb&1] and all
        // waves are done reading the buffer we are about to refill.
        asm volatile("s_wait_asynccnt 0x0" ::: "memory");
        __syncthreads();
        if (cb < 7) issue(cb + 1, (cb + 1) & 1);       // overlaps WMMA below

        const int ci0 = cb * 32;
        const _Float16* xbuf = xs[cb & 1];

        for (int r = 0; r < 9; ++r) {                  // 9 filter taps
            const int fy = r / 3;
            const int fx = r % 3;

            // A fragments: contiguous k order => lanes>=16 hold K+16 (matches B)
            v16h afrag[2];
            for (int ms = 0; ms < 2; ++ms) {
                int co = mt * 64 + wm * 32 + ms * 16 + l16;
                const _Float16* ap = Amod
                    + (size_t)(b * C_OUT + co) * KTOT + r * 256 + ci0 + lhi * 16;
                v16h a = *(const v16h*)ap;
                afrag[ms] = a;
            }

            // B fragments: contiguous k order (matches A) => lanes>=16 hold K+16;
            // ci innermost => 16 contiguous halves per lane (32B aligned)
            v16h bfrag[2];
            for (int ns = 0; ns < 2; ++ns) {
                int n    = nt * 128 + wn * 32 + ns * 16 + l16;
                int h    = n >> 6;
                int ww   = n & 63;
                int rowi = (h - h0) + fy;          // 0..3
                int coli = ww + fx;                // 0..65
                bfrag[ns] = *(const v16h*)&xbuf[(rowi * 66 + coli) * 32 + lhi * 16];
            }

            for (int ms = 0; ms < 2; ++ms)
                for (int ns = 0; ns < 2; ++ns)
                    acc[ms][ns] = __builtin_amdgcn_wmma_f32_16x16x32_f16(
                        false, afrag[ms], false, bfrag[ns],
                        (short)0, acc[ms][ns], false, false);
        }
    }

    // ---- epilogue: C/D layout VGPR v -> M = v + 8*lhi, N = l16 ----
    for (int ms = 0; ms < 2; ++ms) {
        int cobase = mt * 64 + wm * 32 + ms * 16 + lhi * 8;
        for (int ns = 0; ns < 2; ++ns) {
            int n = nt * 128 + wn * 32 + ns * 16 + l16;
            for (int v = 0; v < 8; ++v) {
                int co = cobase + v;
                out[(size_t)(b * C_OUT + co) * NPIX + n] = acc[ms][ns][v] + bias[co];
            }
        }
    }
}

// ---------------------------------------------------------------------------
// Host-side launch
// ---------------------------------------------------------------------------
extern "C" void kernel_launch(void* const* d_in, const int* in_sizes, int n_in,
                              void* d_out, int out_size, void* d_ws, size_t ws_size,
                              hipStream_t stream)
{
    const float* x    = (const float*)d_in[0];   // [8,256,64,64]
    const float* s    = (const float*)d_in[1];   // [8,512]
    const float* w    = (const float*)d_in[2];   // [256,256,3,3]
    const float* bias = (const float*)d_in[3];   // [256,1,1]
    const float* aw   = (const float*)d_in[4];   // [256,512]
    const float* ab   = (const float*)d_in[5];   // [256]
    float* out        = (float*)d_out;           // [8,256,64,64]

    char* ws = (char*)d_ws;
    float*    style1 = (float*)ws;                                   // 8,192 B
    _Float16* Amod   = (_Float16*)(ws + 8192);                       // 9,437,184 B
    _Float16* Xh     = (_Float16*)(ws + 8192 + (size_t)BS * C_OUT * KTOT * 2);
                                                                     // 16,777,216 B

    style_kernel<<<8, 256, 0, stream>>>(s, aw, ab, style1);
    modw_kernel<<<BS * C_OUT, 256, 0, stream>>>(w, style1, Amod);
    xcvt_kernel<<<(BS * C_IN * NPIX) / (64 * 32), 256, 0, stream>>>(x, Xh);

    dim3 grid(32, 4, BS);
    modconv_gemm<<<grid, 256, 0, stream>>>(Amod, Xh, bias, out);
}
